// LocalMIDIGenerator_78357383348438
// MI455X (gfx1250) — matrix operand order repaired and checked by reading, not compiled, not run on hardware
//
#include <hip/hip_runtime.h>
#include <hip/hip_bf16.h>
#include <math.h>

// ---------------- model constants ----------------
#define BB 2
#define LL 1024
#define DD 768
#define DIN 1536
#define NSTATE 16
#define DT_RANK 48
#define D_CONV 4
#define VOCABSZ 512
#define NLAYERS 2
#define MROWS (BB * LL)          // 2048
#define XDBL_W (DT_RANK + 2 * NSTATE)   // 80

typedef __attribute__((ext_vector_type(16))) _Float16 v16h;
typedef __attribute__((ext_vector_type(8)))  _Float16 v8h;
typedef __attribute__((ext_vector_type(8)))  float    v8f;

__constant__ float c_freqs[4] = {6.28318530718f, 3.14159265359f,
                                 1.57079632679f, 0.78539816340f};

__device__ __forceinline__ float silu_f(float x) {
    return x / (1.0f + __expf(-x));
}

// -------- weight convert: W (Nout,K) f32 row-major -> Wt (Kpad,Npad) f16, zero padded --------
// Npad is a multiple of 64 so a wave's full 16x64 strip is always in-bounds.
__global__ void convert_wt_kernel(const float* __restrict__ W, _Float16* __restrict__ Wt,
                                  int Nout, int K, int Npad, int Kpad) {
    int i = blockIdx.x * 256 + threadIdx.x;
    int tot = Kpad * Npad;
    if (i >= tot) return;
    int k = i / Npad;
    int n = i - k * Npad;
    float v = (k < K && n < Nout) ? W[n * K + k] : 0.0f;
    Wt[i] = (_Float16)v;
}

// -------- f32 -> f16 activation convert --------
__global__ void convert_act_kernel(const float* __restrict__ src, _Float16* __restrict__ dst,
                                   int n) {
    int i = blockIdx.x * 256 + threadIdx.x;
    if (i < n) dst[i] = (_Float16)src[i];
}

__device__ __forceinline__ void store_tile(v8f c, int m0, int n0, int r16, int hi,
                                           const float* bias, float* __restrict__ C, int ldc,
                                           _Float16* __restrict__ C16, int ldc16,
                                           int Nreal, int act) {
    int col = n0 + r16;
    if (col >= Nreal) return;
    float bv = bias ? bias[col] : 0.0f;
    #pragma unroll
    for (int r = 0; r < 8; ++r) {
        int row = m0 + r + hi * 8;
        float v = c[r] + bv;
        if (act == 1)      v = fmaxf(v, 0.0f);
        else if (act == 2) v = (v > 20.0f) ? v : log1pf(__expf(v));
        C[(size_t)row * ldc + col] = v;
        if (C16) C16[(size_t)row * ldc16 + col] = (_Float16)v;
    }
}

// -------- WMMA GEMM: C = act( A16(M,Kpad;lda) . Wt^T + bias ) --------
// A16: f16 activations (row-major, lda halves). Wt: (Kpad,Npad) f16 row-major = B[k][n],
// zero padded in both K (annihilates stale A columns) and N (64-wide wave strips always valid).
// Block = 4 waves. Each wave computes a 16x64 output strip (4 WMMA accumulators) sharing a
// 16xKpad A tile that is DMA'd into LDS with async global->LDS loads (ASYNCcnt).
__global__ void wmma_gemm_kernel(const _Float16* __restrict__ A16, int lda,
                                 const _Float16* __restrict__ Wt, int ldb /*Npad, %64==0*/,
                                 const float* __restrict__ bias,
                                 float* __restrict__ C, int ldc,
                                 _Float16* __restrict__ C16, int ldc16,
                                 int Kpad, int Nreal, int act) {
    extern __shared__ char smem_raw[];
    _Float16* smem = (_Float16*)smem_raw;
    const uint32_t lds_base = (uint32_t)(uintptr_t)smem;   // low 32 bits = LDS byte address

    const int m0  = blockIdx.x * 16;
    const int tid = threadIdx.y * 32 + threadIdx.x;

    // ---- async DMA of the A tile (16 rows x Kpad halves) into LDS ----
    const int cpr    = Kpad / 8;        // 16-byte chunks per row
    const int chunks = 16 * cpr;
    for (int c = tid; c < chunks; c += 128) {
        int r  = c / cpr;
        int cc = c - r * cpr;
        const _Float16* g = A16 + (size_t)(m0 + r) * lda + cc * 8;
        uint32_t lo = lds_base + (uint32_t)(r * Kpad + cc * 8) * 2u;
        asm volatile("global_load_async_to_lds_b128 %0, %1, off"
                     :: "v"(lo), "v"(g) : "memory");
    }
    asm volatile("s_wait_asynccnt 0x0" ::: "memory");
    __syncthreads();

    const int lane = threadIdx.x;
    const int hi   = lane >> 4;        // lane group (0: lanes 0-15, 1: lanes 16-31)
    const int r16  = lane & 15;
    const int n0w  = blockIdx.y * 256 + threadIdx.y * 64;
    if (n0w >= ldb) return;            // fully padded wave: idle (after barrier)

    v8f c0 = {}, c1 = {}, c2 = {}, c3 = {};
    union AFrag { v16h v; v8h h[2]; } a;

    for (int kb = 0; kb < Kpad; kb += 32) {
        // A fragment per ISA 16-bit 16x32 layout:
        // lanes 0-15: halves 0-7 -> K=kb+0..7,  halves 8-15 -> K=kb+16..23
        // lanes16-31: halves 0-7 -> K=kb+8..15, halves 8-15 -> K=kb+24..31
        const _Float16* ap = smem + r16 * Kpad + kb + hi * 8;
        a.h[0] = *(const v8h*)(ap);
        a.h[1] = *(const v8h*)(ap + 16);

        // B fragments: lane holds N column r16 (per 16-wide tile); halves follow the same
        // K order as A (0-7 -> kb+8*hi+0..7, 8-15 -> kb+16+8*hi+0..7).
        int krow = kb + r16 + hi * 16;
        const _Float16* wrow = Wt + (size_t)krow * ldb + n0w;
        __builtin_prefetch((const void*)(wrow + (size_t)32 * ldb), 0, 1);

        v16h b0, b1, b2, b3;
        #pragma unroll
        for (int s = 0; s < 16; ++s) {
            int kk = kb + hi * 8 + (s < 8 ? s : 8 + s);
            const _Float16* bp = Wt + (size_t)kk * ldb + n0w + r16;
            b0[s] = bp[0];
            b1[s] = bp[16];
            b2[s] = bp[32];
            b3[s] = bp[48];
        }
        c0 = __builtin_amdgcn_wmma_f32_16x16x32_f16(false, a.v, false, b0, (short)0, c0, false, false);
        c1 = __builtin_amdgcn_wmma_f32_16x16x32_f16(false, a.v, false, b1, (short)0, c1, false, false);
        c2 = __builtin_amdgcn_wmma_f32_16x16x32_f16(false, a.v, false, b2, (short)0, c2, false, false);
        c3 = __builtin_amdgcn_wmma_f32_16x16x32_f16(false, a.v, false, b3, (short)0, c3, false, false);
    }

    store_tile(c0, m0, n0w,      r16, hi, bias, C, ldc, C16, ldc16, Nreal, act);
    store_tile(c1, m0, n0w + 16, r16, hi, bias, C, ldc, C16, ldc16, Nreal, act);
    store_tile(c2, m0, n0w + 32, r16, hi, bias, C, ldc, C16, ldc16, Nreal, act);
    store_tile(c3, m0, n0w + 48, r16, hi, bias, C, ldc, C16, ldc16, Nreal, act);
}

// -------- embedding sum + rotary (writes f16 activation for the GEMMs) --------
__global__ void assemble_rotary_kernel(const int* __restrict__ midi,
                                       const int* __restrict__ tempo,
                                       const float* __restrict__ Em,
                                       const float* __restrict__ Et,
                                       const float* __restrict__ cbuf,
                                       _Float16* __restrict__ x16) {
    int idx = blockIdx.x * 256 + threadIdx.x;           // over B*L*(D/2)
    if (idx >= MROWS * (DD / 2)) return;
    int i  = idx % (DD / 2);
    int bl = idx / (DD / 2);
    int tm = midi[bl];
    int tt = tempo[bl];
    int j0 = 2 * i, j1 = 2 * i + 1;
    float x1 = Em[tm * DD + j0] + Et[tt * DD + j0] + cbuf[(size_t)bl * DD + j0];
    float x2 = Em[tm * DD + j1] + Et[tt * DD + j1] + cbuf[(size_t)bl * DD + j1];
    int l = bl % LL;
    int g = i / 96;                                     // freq group
    float ang = (float)l * c_freqs[g];
    float s = sinf(ang), cc = cosf(ang);
    x16[(size_t)bl * DD + i]          = (_Float16)(x1 * cc - x2 * s);
    x16[(size_t)bl * DD + DD / 2 + i] = (_Float16)(x2 * cc + x1 * s);
}

// -------- depthwise causal conv (D_CONV=4) + SiLU --------
__global__ void conv_silu_kernel(const float* __restrict__ xz,
                                 const float* __restrict__ cw,
                                 const float* __restrict__ cb,
                                 float* __restrict__ xc,
                                 _Float16* __restrict__ xc16) {
    int idx = blockIdx.x * 256 + threadIdx.x;           // B*L*DIN
    if (idx >= MROWS * DIN) return;
    int d  = idx % DIN;
    int bl = idx / DIN;
    int l  = bl % LL;
    int b  = bl / LL;
    float acc = cb[d];
    #pragma unroll
    for (int k = 0; k < D_CONV; ++k) {
        int lp = l + k - (D_CONV - 1);
        if (lp >= 0)
            acc += xz[((size_t)(b * LL + lp)) * (2 * DIN) + d] * cw[d * D_CONV + k];
    }
    float v = silu_f(acc);
    xc[idx]   = v;
    xc16[idx] = (_Float16)v;
}

// -------- selective scan + skip + SiLU gating; one thread per (b,d) channel --------
__global__ void scan_kernel(const float* __restrict__ dtb,
                            const float* __restrict__ xc,
                            const float* __restrict__ xdbl,
                            const float* __restrict__ xz,
                            const float* __restrict__ A_log,
                            const float* __restrict__ Dp,
                            _Float16* __restrict__ y16) {
    int t = blockIdx.x * blockDim.x + threadIdx.x;      // B*DIN
    if (t >= BB * DIN) return;
    int d = t % DIN;
    int b = t / DIN;
    float a[NSTATE], h[NSTATE];
    #pragma unroll
    for (int n = 0; n < NSTATE; ++n) {
        a[n] = -__expf(A_log[d * NSTATE + n]);
        h[n] = 0.0f;
    }
    float dpar = Dp[d];
    for (int l = 0; l < LL; ++l) {
        size_t bl = (size_t)b * LL + l;
        float dt = dtb[bl * DIN + d];
        float xv = xc[bl * DIN + d];
        const float* xr = xdbl + bl * XDBL_W;
        float dx = dt * xv;
        float acc = 0.0f;
        #pragma unroll
        for (int n = 0; n < NSTATE; ++n) {
            float Bn = xr[DT_RANK + n];
            float Cn = xr[DT_RANK + NSTATE + n];
            h[n] = h[n] * __expf(dt * a[n]) + dx * Bn;
            acc += h[n] * Cn;
        }
        acc += dpar * xv;
        float z = xz[bl * (2 * DIN) + DIN + d];
        acc *= silu_f(z);
        y16[bl * DIN + d] = (_Float16)acc;
    }
}

// -------- tempo head GEMV: out[bl] = dot(H[bl,:], w) + b --------
__global__ void tempo_gemv_kernel(const float* __restrict__ H,
                                  const float* __restrict__ w,
                                  const float* __restrict__ b2,
                                  float* __restrict__ out) {
    int bl = blockIdx.x;
    int lane = threadIdx.x;
    float s = 0.0f;
    for (int k = lane; k < DD; k += 32)
        s += H[(size_t)bl * DD + k] * w[k];
    #pragma unroll
    for (int off = 16; off > 0; off >>= 1)
        s += __shfl_down(s, off, 32);
    if (lane == 0) out[bl] = s + b2[0];
}

// ---------------- host side ----------------
static inline int rup(int x, int m) { return (x + m - 1) / m * m; }

// workspace layout: f32 region then f16 region (all offsets 16B+ aligned)
#define OFF_XZ    0            // 6291456 f32  (xz; also tempo/rhythm-hidden f32 sink)
#define OFF_XC    6291456      // 3145728 f32  (xc; also control-embed buffer pre-layer)
#define OFF_DT    9437184      // 3145728 f32
#define OFF_XDBL  12582912     // 163840  f32
#define OFF_XF    12746752     // 1572864 f32  (f32 sink for out_proj C)
#define OFF_H16F  14319616     // f16 arena starts here (float units)
// f16 offsets (half units) within arena:
#define H_X16     0            // 1572864
#define H_XC16    1572864      // 3145728
#define H_XDBL16  4718592      // 163840
#define H_Y16     4882432      // 3145728
#define H_CT16    8028160      // 1572864
#define H_HH16    9601024      // 1572864
#define H_WT      11173888     // 2621440

extern "C" void kernel_launch(void* const* d_in, const int* in_sizes, int n_in,
                              void* d_out, int out_size, void* d_ws, size_t ws_size,
                              hipStream_t stream) {
    const int*   midi   = (const int*)  d_in[0];
    const float* ctrl   = (const float*)d_in[1];
    const int*   tempoi = (const int*)  d_in[2];
    const float* E_midi = (const float*)d_in[3];
    const float* E_temp = (const float*)d_in[4];
    const float* Wc     = (const float*)d_in[5];
    const float* bc     = (const float*)d_in[6];
    const float* in_w   = (const float*)d_in[7];
    const float* conv_w = (const float*)d_in[8];
    const float* conv_b = (const float*)d_in[9];
    const float* xp_w   = (const float*)d_in[10];
    const float* dtp_w  = (const float*)d_in[11];
    const float* dt_b   = (const float*)d_in[12];
    const float* A_log  = (const float*)d_in[13];
    const float* D_par  = (const float*)d_in[14];
    const float* out_w  = (const float*)d_in[15];
    const float* Wo     = (const float*)d_in[16];
    const float* bo     = (const float*)d_in[17];
    const float* tw1    = (const float*)d_in[18];
    const float* tb1    = (const float*)d_in[19];
    const float* tw2    = (const float*)d_in[20];
    const float* tb2    = (const float*)d_in[21];
    const float* rw1    = (const float*)d_in[22];
    const float* rb1    = (const float*)d_in[23];
    const float* rw2    = (const float*)d_in[24];
    const float* rb2    = (const float*)d_in[25];

    float* ws   = (float*)d_ws;
    float* xz   = ws + OFF_XZ;
    float* xc   = ws + OFF_XC;     // also control-embed buffer pre-layer
    float* dt   = ws + OFF_DT;
    float* xdbl = ws + OFF_XDBL;
    float* xf   = ws + OFF_XF;
    _Float16* hb     = (_Float16*)(ws + OFF_H16F);
    _Float16* x16    = hb + H_X16;
    _Float16* xc16   = hb + H_XC16;
    _Float16* xdbl16 = hb + H_XDBL16;
    _Float16* y16    = hb + H_Y16;
    _Float16* ct16   = hb + H_CT16;
    _Float16* hh16   = hb + H_HH16;
    _Float16* wt     = hb + H_WT;

    float* out_logits = (float*)d_out;                  // (B,L,512)
    float* out_rhythm = out_logits + MROWS * VOCABSZ;   // (B,L,4)
    float* out_tempo  = out_rhythm + MROWS * 4;         // (B,L)

    auto gemm = [&](const _Float16* A16, int lda, const float* W, int Nout, int K,
                    const float* bias, float* Cp, int ldc,
                    _Float16* C16, int ldc16, int act) {
        int Kpad = rup(K, 32);
        int Npad = rup(Nout, 64);   // wave-strip aligned: all 4 WMMA tiles unconditional
        int tot  = Kpad * Npad;
        convert_wt_kernel<<<(tot + 255) / 256, 256, 0, stream>>>(W, wt, Nout, K, Npad, Kpad);
        dim3 grid(MROWS / 16, (Npad + 255) / 256);
        dim3 blk(32, 4);
        size_t smem = (size_t)16 * Kpad * sizeof(_Float16);
        wmma_gemm_kernel<<<grid, blk, smem, stream>>>(A16, lda, wt, Npad, bias, Cp, ldc,
                                                      C16, ldc16, Kpad, Nout, act);
    };

    // --- pre-layer: control embed, tempo head, embedding sum + rotary ---
    {
        int n = MROWS * DD;
        convert_act_kernel<<<(n + 255) / 256, 256, 0, stream>>>(ctrl, ct16, n);
    }
    gemm(ct16, DD, Wc, DD, DD, bc, xc, DD, nullptr, 0, 0);        // cbuf = ctrl @ Wc^T + bc
    gemm(ct16, DD, tw1, DD, DD, tb1, xz, DD, nullptr, 0, 1);      // Hbuf = relu(ctrl@tw1^T+tb1)
    tempo_gemv_kernel<<<MROWS, 32, 0, stream>>>(xz, tw2, tb2, out_tempo);
    {
        int tot = MROWS * (DD / 2);
        assemble_rotary_kernel<<<(tot + 255) / 256, 256, 0, stream>>>(
            midi, tempoi, E_midi, E_temp, xc, x16);
    }

    // --- Mamba layers ---
    for (int i = 0; i < NLAYERS; ++i) {
        const float* in_wi  = in_w  + (size_t)i * 2 * DIN * DD;
        const float* cw_i   = conv_w + (size_t)i * DIN * D_CONV;
        const float* cb_i   = conv_b + (size_t)i * DIN;
        const float* xp_wi  = xp_w  + (size_t)i * XDBL_W * DIN;
        const float* dtp_wi = dtp_w + (size_t)i * DIN * DT_RANK;
        const float* dtb_i  = dt_b  + (size_t)i * DIN;
        const float* Al_i   = A_log + (size_t)i * DIN * NSTATE;
        const float* Dp_i   = D_par + (size_t)i * DIN;
        const float* ow_i   = out_w + (size_t)i * DD * DIN;

        gemm(x16, DD, in_wi, 2 * DIN, DD, nullptr, xz, 2 * DIN, nullptr, 0, 0);  // xz
        {
            int tot = MROWS * DIN;
            conv_silu_kernel<<<(tot + 255) / 256, 256, 0, stream>>>(xz, cw_i, cb_i, xc, xc16);
        }
        gemm(xc16, DIN, xp_wi, XDBL_W, DIN, nullptr, xdbl, XDBL_W, xdbl16, XDBL_W, 0);
        gemm(xdbl16, XDBL_W, dtp_wi, DIN, DT_RANK, dtb_i, dt, DIN, nullptr, 0, 2); // softplus
        {
            int tot = BB * DIN;
            scan_kernel<<<(tot + 255) / 256, 256, 0, stream>>>(
                dt, xc, xdbl, xz, Al_i, Dp_i, y16);
        }
        gemm(y16, DIN, ow_i, DD, DIN, nullptr, xf, DD, x16, DD, 0);  // x = y @ out_w^T
    }

    // --- heads ---
    gemm(x16, DD, Wo, VOCABSZ, DD, bo, out_logits, VOCABSZ, nullptr, 0, 0);   // logits
    gemm(x16, DD, rw1, DD, DD, rb1, xz, DD, hh16, DD, 1);                     // rhythm hidden
    gemm(hh16, DD, rw2, 4, DD, rb2, out_rhythm, 4, nullptr, 0, 0);            // rhythm logits

    (void)in_sizes; (void)n_in; (void)out_size; (void)ws_size;
}
